// MultiHeadedAttention_21672404975995
// MI455X (gfx1250) — matrix operand order repaired and checked by reading, not compiled, not run on hardware
//
#include <hip/hip_runtime.h>
#include <hip/hip_bf16.h>
#include <cmath>

// ---------------------------------------------------------------------------
// MultiHeadedAttention for MI455X (gfx1250, wave32, WMMA f16 16x16x32)
// B=2, S=2048, D=1024, H=16, DK=64
// ---------------------------------------------------------------------------

typedef _Float16 half_t;
typedef __attribute__((ext_vector_type(16))) _Float16 v16h;
typedef __attribute__((ext_vector_type(8)))  _Float16 h8;
typedef __attribute__((ext_vector_type(8)))  float    v8f;
typedef int i32x4 __attribute__((vector_size(16)));

constexpr int Bb  = 2;
constexpr int Ss  = 2048;
constexpr int Dd  = 1024;
constexpr int Hh  = 16;
constexpr int DKk = 64;
constexpr int Mm  = Bb * Ss;   // 4096 rows in every GEMM

// ---- async global->LDS staging (CDNA5 GLOBAL_LOAD_ASYNC_TO_LDS_B128) ------
#if defined(__HIP_DEVICE_COMPILE__) && \
    __has_builtin(__builtin_amdgcn_global_load_async_to_lds_b128)
#define ASYNC_LDS 1
#else
#define ASYNC_LDS 0
#endif

__device__ __forceinline__ void copy16_async(const half_t* gsrc, half_t* ldst) {
#if ASYNC_LDS
  typedef __attribute__((address_space(1))) i32x4* gp_t;
  typedef __attribute__((address_space(3))) i32x4* lp_t;
  // generic LDS pointer low 32 bits == wave-relative LDS byte address
  __builtin_amdgcn_global_load_async_to_lds_b128(
      (gp_t)(uintptr_t)gsrc, (lp_t)(uint32_t)(uintptr_t)ldst, 0, 0);
#else
  *(h8*)ldst = *(const h8*)gsrc;
#endif
}

__device__ __forceinline__ void copy_wait() {
#if ASYNC_LDS
#if __has_builtin(__builtin_amdgcn_s_wait_asynccnt)
  __builtin_amdgcn_s_wait_asynccnt(0);
#else
  asm volatile("s_wait_asynccnt 0" ::: "memory");
#endif
#endif
}

// Build a 16-half WMMA operand from two contiguous 8-half LDS chunks.
__device__ __forceinline__ v16h frag2(const half_t* p0, const half_t* p1) {
  h8 lo = *(const h8*)p0;
  h8 hi = *(const h8*)p1;
  v16h r;
#pragma unroll
  for (int i = 0; i < 8; ++i) { r[i] = lo[i]; r[i + 8] = hi[i]; }
  return r;
}

__device__ __forceinline__ v8f wmma_f16(v16h a, v16h b, v8f c) {
  // D = A(16x32 f16) * B(32x16 f16) + C(16x16 f32)
  return __builtin_amdgcn_wmma_f32_16x16x32_f16(false, a, false, b,
                                                (short)0, c, false, false);
}

// ---------------------------------------------------------------------------
// GEMM: Out[m][n] = sum_k X[m][k] * W[n][k] + bias[n]   (torch Linear)
// M=4096, N=1024, K=1024. 128x64 block tile, 8 waves, 16x64 C per wave
// (1 A-frag + 4 B-frags -> 4 WMMAs per K-step per wave).
// MODE 0: f16 output, [b][h][s][dk] head layout        (Q, K)
// MODE 2: f16 output, [b][h][dk][s] transposed layout  (V)
// MODE 1: f32 output, row-major [m][n]                 (final -> d_out)
// ---------------------------------------------------------------------------
template <typename TX, int MODE>
__global__ __launch_bounds__(256) void gemm64(const TX* __restrict__ X,
                                              const float* __restrict__ W,
                                              const float* __restrict__ bias,
                                              void* __restrict__ OutP) {
  constexpr int SA = 40;  // padded LDS row stride (halfs): 80B, 16B-aligned chunks
  __shared__ alignas(16) half_t At[128 * SA];
  __shared__ alignas(16) half_t Bt[64 * SA];

  const int tid  = threadIdx.x;
  const int wave = tid >> 5, lane = tid & 31;
  const int j = lane & 15, hi = lane >> 4;
  const int m0 = blockIdx.y * 128, n0 = blockIdx.x * 64;

  // staging coordinates
  const int arow = (tid * 16) >> 5, akk = (tid * 16) & 31;  // A: 16 elems/thread
  const int brow = (tid * 8) >> 5,  bkk = (tid * 8) & 31;   // B: 8 elems/thread

  v8f c[4] = {{}, {}, {}, {}};

  for (int k0 = 0; k0 < Dd; k0 += 32) {
    // Stage A (128x32 of X) and B (64x32 of W rows) as f16 in LDS.
    {
      const TX* src = X + (size_t)(m0 + arow) * Dd + k0 + akk;
      half_t* dst = At + arow * SA + akk;
#pragma unroll
      for (int i = 0; i < 16; ++i) dst[i] = (half_t)(float)src[i];
      const float* wsrc = W + (size_t)(n0 + brow) * Dd + k0 + bkk;
      half_t* wdst = Bt + brow * SA + bkk;
#pragma unroll
      for (int i = 0; i < 8; ++i) wdst[i] = (half_t)wsrc[i];
      if (k0 + 32 < Dd) {  // prefetch next K tile (global_prefetch_b8)
        __builtin_prefetch(src + 32, 0, 3);
        __builtin_prefetch(wsrc + 32, 0, 3);
      }
    }
    __syncthreads();

    // A fragment: wave's 16-row stripe, K=0..31 of this tile
    const half_t* ar = At + (wave * 16 + j) * SA;
    const v16h a = frag2(ar + hi * 16, ar + hi * 16 + 8);
    // B fragments: W row-major == B(KxN) column-major -> contiguous loads
#pragma unroll
    for (int g = 0; g < 4; ++g) {
      const half_t* bp = Bt + (g * 16 + j) * SA + hi * 16;
      c[g] = wmma_f16(a, frag2(bp, bp + 8), c[g]);
    }
    __syncthreads();
  }

  // Epilogue: bias + store. Lane holds rows m0+wave*16+hi*8+r, cols n0+g*16+j.
#pragma unroll
  for (int g = 0; g < 4; ++g) {
    const int col = n0 + g * 16 + j;
    const float bv = bias[col];
#pragma unroll
    for (int r = 0; r < 8; ++r) {
      const int m = m0 + wave * 16 + hi * 8 + r;
      const float v = c[g][r] + bv;
      if constexpr (MODE == 0) {
        half_t* Out = (half_t*)OutP;
        const int b = m >> 11, s = m & (Ss - 1);
        const int h = col >> 6, d = col & 63;
        Out[((size_t)(b * Hh + h) * Ss + s) * DKk + d] = (half_t)v;
      } else if constexpr (MODE == 2) {
        half_t* Out = (half_t*)OutP;
        const int b = m >> 11, s = m & (Ss - 1);
        const int h = col >> 6, d = col & 63;
        Out[((size_t)(b * Hh + h) * DKk + d) * Ss + s] = (half_t)v;
      } else {
        float* Out = (float*)OutP;
        Out[(size_t)m * Dd + col] = v;
      }
    }
  }
}

// ---------------------------------------------------------------------------
// Flash attention, causal. One workgroup = 128 query rows of one (b,h).
// 8 waves x 16 rows; K-blocks of 32; online softmax; all matmuls via WMMA.
// K/V tiles staged with async global->LDS copies (V arrives pre-transposed).
// ---------------------------------------------------------------------------
__global__ __launch_bounds__(256) void attn_kernel(const half_t* __restrict__ Qh,
                                                   const half_t* __restrict__ Kh,
                                                   const half_t* __restrict__ Vh,
                                                   half_t* __restrict__ Xb) {
  constexpr int SQ = 72, SK = 72, SV = 40, SP = 40;
  __shared__ alignas(16) half_t Qs[128 * SQ];
  __shared__ alignas(16) half_t Ks[32 * SK];
  __shared__ alignas(16) half_t Vt[64 * SV];       // [d][n], fed from Vh[b][h][d][s]
  __shared__ alignas(16) half_t Ps[8 * 16 * SP];   // per-wave P scratch

  const int tid = threadIdx.x, wave = tid >> 5, lane = tid & 31;
  const int j = lane & 15, hi = lane >> 4;
  const int qblk = blockIdx.x & 15;   // S/128 = 16 query blocks
  const int bh   = blockIdx.x >> 4;   // 0..31 (b*H + h)
  const int qBase = qblk * 128;
  const size_t headBase = (size_t)bh * Ss * DKk;    // Q/K head layout
  const int bIdx = bh >> 4, hIdx = bh & 15;

  // Stage Q tile (128x64), pre-scaled by 1/sqrt(DK)=0.125 (exact in f16)
  for (int c = tid; c < 128 * 8; c += 256) {
    const int row = c >> 3, d = (c & 7) * 8;
    h8 v = *(const h8*)(Qh + headBase + (size_t)(qBase + row) * DKk + d);
#pragma unroll
    for (int i = 0; i < 8; ++i) v[i] = v[i] * (half_t)0.125f;
    *(h8*)(Qs + row * SQ + d) = v;
  }
  __syncthreads();

  // Q A-fragments (d = 0..31 and 32..63) held in registers for the whole loop
  const half_t* qrow = Qs + (wave * 16 + j) * SQ;
  const v16h qa0 = frag2(qrow + hi * 16, qrow + hi * 16 + 8);
  const v16h qa1 = frag2(qrow + 32 + hi * 16, qrow + 40 + hi * 16);

  v8f o0 = {}, o1 = {}, o2 = {}, o3 = {};
  float mrow[8], lrow[8];
#pragma unroll
  for (int r = 0; r < 8; ++r) { mrow[r] = -INFINITY; lrow[r] = 0.f; }

  const int qRow0 = qBase + wave * 16;
  const int nkb = (qblk + 1) * 4;  // causal: blocks through max q of workgroup

  for (int kb = 0; kb < nkb; ++kb) {
    const int k0 = kb * 32;
    // Stage K tile (32x64, row-major): 256 async b128 chunks, 1 per thread
    {
      const int row = tid >> 3, d = (tid & 7) * 8;
      copy16_async(Kh + headBase + (size_t)(k0 + row) * DKk + d,
                   Ks + row * SK + d);
    }
    // Stage Vt tile (64x32): Vh is pre-transposed [b][h][d][s] -> contiguous
    {
      const int d = tid >> 2, part = (tid & 3) * 8;
      copy16_async(Vh + ((size_t)bh * DKk + d) * Ss + k0 + part,
                   Vt + d * SV + part);
    }
    copy_wait();
    __syncthreads();

    if (k0 <= qRow0 + 15) {  // wave-uniform causal skip; EXEC stays full
      // scores S(16x32) = Q(16x64) * K^T : 4 WMMAs
      v8f s0 = {}, s1 = {};
      {
        const half_t* kp = Ks + j * SK;
        s0 = wmma_f16(qa0, frag2(kp + hi * 16, kp + hi * 16 + 8), s0);
        s0 = wmma_f16(qa1, frag2(kp + 32 + hi * 16, kp + 40 + hi * 16), s0);
        kp = Ks + (16 + j) * SK;
        s1 = wmma_f16(qa0, frag2(kp + hi * 16, kp + hi * 16 + 8), s1);
        s1 = wmma_f16(qa1, frag2(kp + 32 + hi * 16, kp + 40 + hi * 16), s1);
      }
      // causal mask in C-frag layout: row m = r + 8*hi, col n = j (+16 for s1)
      const int col0 = k0 + j, col1 = k0 + 16 + j;
#pragma unroll
      for (int r = 0; r < 8; ++r) {
        const int q = qRow0 + hi * 8 + r;
        if (col0 > q) s0[r] = -1e30f;
        if (col1 > q) s1[r] = -1e30f;
      }
      // online softmax: 16-lane row reductions (stay within lane half)
#pragma unroll
      for (int r = 0; r < 8; ++r) {
        float mx = fmaxf(s0[r], s1[r]);
#pragma unroll
        for (int msk = 1; msk < 16; msk <<= 1)
          mx = fmaxf(mx, __shfl_xor(mx, msk, 32));
        const float mnew = fmaxf(mrow[r], mx);
        const float sc = __expf(mrow[r] - mnew);
        const float p0 = __expf(s0[r] - mnew);
        const float p1 = __expf(s1[r] - mnew);
        float rs = p0 + p1;
#pragma unroll
        for (int msk = 1; msk < 16; msk <<= 1)
          rs += __shfl_xor(rs, msk, 32);
        lrow[r] = lrow[r] * sc + rs;
        mrow[r] = mnew;
        o0[r] *= sc; o1[r] *= sc; o2[r] *= sc; o3[r] *= sc;
        // write P (f16) to per-wave scratch for C-frag -> A-frag relayout
        half_t* pw = Ps + wave * (16 * SP) + (hi * 8 + r) * SP;
        pw[j] = (half_t)p0;
        pw[16 + j] = (half_t)p1;
      }
      // per-wave LDS store->load dependency across lanes
      asm volatile("s_wait_dscnt 0" ::: "memory");

      const half_t* prow = Ps + wave * (16 * SP) + j * SP;
      const v16h pa = frag2(prow + hi * 16, prow + hi * 16 + 8);
      // O(16x64) += P(16x32) * V(32x64) : 4 WMMAs against transposed V
      const half_t* vp = Vt + j * SV + hi * 16;
      o0 = wmma_f16(pa, frag2(vp, vp + 8), o0);
      vp = Vt + (16 + j) * SV + hi * 16;
      o1 = wmma_f16(pa, frag2(vp, vp + 8), o1);
      vp = Vt + (32 + j) * SV + hi * 16;
      o2 = wmma_f16(pa, frag2(vp, vp + 8), o2);
      vp = Vt + (48 + j) * SV + hi * 16;
      o3 = wmma_f16(pa, frag2(vp, vp + 8), o3);
    }
    __syncthreads();
  }

  // normalize and store f16 to [b][s][h*64+d] for the output projection GEMM
#pragma unroll
  for (int r = 0; r < 8; ++r) {
    const float inv = 1.0f / lrow[r];
    const int q = qRow0 + hi * 8 + r;
    half_t* dst = Xb + ((size_t)bIdx * Ss + q) * Dd + hIdx * DKk;
    dst[j]      = (half_t)(o0[r] * inv);
    dst[16 + j] = (half_t)(o1[r] * inv);
    dst[32 + j] = (half_t)(o2[r] * inv);
    dst[48 + j] = (half_t)(o3[r] * inv);
  }
}

// ---------------------------------------------------------------------------
extern "C" void kernel_launch(void* const* d_in, const int* in_sizes, int n_in,
                              void* d_out, int out_size, void* d_ws, size_t ws_size,
                              hipStream_t stream) {
  (void)in_sizes; (void)n_in; (void)out_size; (void)ws_size;
  const float* query = (const float*)d_in[0];
  const float* key_  = (const float*)d_in[1];
  const float* value = (const float*)d_in[2];
  // d_in[3] = causal mask (tril) -- implemented analytically in attn_kernel
  const float* Wq = (const float*)d_in[4];
  const float* bq = (const float*)d_in[5];
  const float* Wk = (const float*)d_in[6];
  const float* bk = (const float*)d_in[7];
  const float* Wv = (const float*)d_in[8];
  const float* bv = (const float*)d_in[9];
  const float* Wo = (const float*)d_in[10];
  const float* bo = (const float*)d_in[11];

  // Workspace: 4 f16 tensors of B*S*D = 4M elements (8 MiB) each
  half_t* ws = (half_t*)d_ws;
  const size_t elems = (size_t)Mm * Dd;
  half_t* Qh = ws;              // [b][h][s][dk]
  half_t* Kh = ws + elems;      // [b][h][s][dk]
  half_t* Vh = ws + 2 * elems;  // [b][h][dk][s]  (pre-transposed for PV WMMA)
  half_t* Xb = ws + 3 * elems;  // [b][s][h*dk]

  dim3 g(Dd / 64, Mm / 128);  // 16 x 32 workgroups
  gemm64<float, 0><<<g, 256, 0, stream>>>(query, Wq, bq, Qh);
  gemm64<float, 0><<<g, 256, 0, stream>>>(key_, Wk, bk, Kh);
  gemm64<float, 2><<<g, 256, 0, stream>>>(value, Wv, bv, Vh);
  attn_kernel<<<Bb * Hh * (Ss / 128), 256, 0, stream>>>(Qh, Kh, Vh, Xb);
  gemm64<half_t, 1><<<g, 256, 0, stream>>>(Xb, Wo, bo, d_out);
}
